// AbstractRouter_67997922231054
// MI455X (gfx1250) — matrix operand order repaired and checked by reading, not compiled, not run on hardware
//
#include <hip/hip_runtime.h>

typedef __attribute__((ext_vector_type(16))) __bf16 v16bf;
typedef __attribute__((ext_vector_type(8)))  float  v8f;
typedef __attribute__((ext_vector_type(4)))  unsigned int u32x4;
typedef __attribute__((ext_vector_type(8)))  int i32x8;
typedef __attribute__((ext_vector_type(4)))  int i32x4;

#define T_TOK   16384
#define D_DIM   2048
#define N_EXP   64
#define K_STEPS (D_DIM / 32)            // 64
#define B_ELTS  (D_DIM * N_EXP)         // 131072 bf16 = 256 KB
#define B_BYTES (B_ELTS * 2)
#define BLOCK_THREADS 512               // 16 waves, 256 rows/block
#define ROWS_PER_BLOCK 256

#if defined(__has_builtin)
#  if __has_builtin(__builtin_amdgcn_tensor_load_to_lds)
#    define HAS_TDM 1
#  endif
#endif

// fp32 -> bf16 (bit form, RNE) for the pack kernel's plain stores
static __device__ __forceinline__ unsigned short f2bfu(float f) {
    unsigned u = __builtin_bit_cast(unsigned, f);
    unsigned r = (u + 0x7FFFu + ((u >> 16) & 1u)) >> 16;
    return (unsigned short)r;
}

// ---------------------------------------------------------------------------
// Kernel 1: pack w_gate [2048 x 64] fp32 into bf16 WMMA B-fragment order.
// packed[((t*4+nb)*32 + lane)*16 + e] = bf16(w_gate[(t*32 + (lane>>4)*8 + (e&7) + (e>>3)*16)*64 + nb*16 + (lane&15)])
// Also zero expert_activation (atomics must start from 0 every call).
// ---------------------------------------------------------------------------
__global__ void router_pack_b(const float* __restrict__ w_gate,
                              unsigned short* __restrict__ packed,
                              float* __restrict__ out_act) {
    int tid = blockIdx.x * blockDim.x + threadIdx.x;          // 0 .. 131071
    int e    = tid & 15;
    int lane = (tid >> 4) & 31;
    int nb   = (tid >> 9) & 3;
    int t    = tid >> 11;
    int K = t * 32 + (lane >> 4) * 8 + (e & 7) + (e >> 3) * 16;
    int N = nb * 16 + (lane & 15);
    packed[tid] = f2bfu(w_gate[(size_t)K * N_EXP + N]);
    if (blockIdx.x == 0 && threadIdx.x < N_EXP) out_act[threadIdx.x] = 0.0f;
}

// ---------------------------------------------------------------------------
// Kernel 2: fused  scores = x@Wg -> +noise -> top2 -> gates -> scatter.
// 512 threads = 16 waves (4/SIMD); wave owns 16 rows x 64 experts.
// B staged once per workgroup into LDS via the Tensor Data Mover.
// ---------------------------------------------------------------------------
__global__ __launch_bounds__(BLOCK_THREADS)
void router_fused(const float* __restrict__ x,
                  const unsigned short* __restrict__ packedB,
                  const float* __restrict__ noise,
                  float* __restrict__ out) {
    extern __shared__ unsigned short smemB[];   // 131072 bf16 = 256 KB

    const int lane   = threadIdx.x & 31;
    const int waveId = threadIdx.x >> 5;
    const int rowbase = blockIdx.x * ROWS_PER_BLOCK + waveId * 16;

    // ---- Stage packed B (global -> LDS) via TDM ----
#if HAS_TDM
    if (threadIdx.x == 0) {
        unsigned long long ga = (unsigned long long)(const void*)packedB;
        unsigned lds_off = (unsigned)(unsigned long long)(size_t)smemB; // LDS byte offset
        // D# group 0: count=1 | lds_addr | global_addr[56:0] | type=2
        u32x4 g0 = { 1u, lds_off, (unsigned)ga,
                     (unsigned)((ga >> 32) & 0x01FFFFFFu) | 0x80000000u };
        // D# group 1: data_size=3 (8B units); 1-D tensor/tile of 32768 units
        i32x8 g1 = { (int)(3u << 16),
                     (int)0x80000000u,      // tensor_dim0 = 32768 (bits 79:48)
                     (int)(1u << 16),       // tensor_dim1 = 1
                     (int)0x80000000u,      // tile_dim0   = 32768 (bits 127:112)
                     0,                     // tile_dim1/2 unused
                     (int)32768,            // tensor_dim0_stride
                     0, 0 };
        i32x4 g2 = { 0, 0, 0, 0 };
        i32x4 g3 = { 0, 0, 0, 0 };
#  if __clang_major__ >= 23
        i32x8 g4 = { 0, 0, 0, 0, 0, 0, 0, 0 };
        __builtin_amdgcn_tensor_load_to_lds(g0, g1, g2, g3, g4, 0);
#  else
        __builtin_amdgcn_tensor_load_to_lds(g0, g1, g2, g3, 0);
#  endif
    }
    __builtin_amdgcn_s_wait_tensorcnt(0);   // no-op in waves that issued nothing
#else
    {   // fallback: cooperative 16B copies
        const uint4* src = (const uint4*)packedB;
        uint4* dst = (uint4*)smemB;
        for (int i = threadIdx.x; i < B_BYTES / 16; i += BLOCK_THREADS) dst[i] = src[i];
    }
#endif
    __syncthreads();

    // ---- GEMM: 16x64 tile in 4 f32 accumulators ----
    v8f acc0 = {}, acc1 = {}, acc2 = {}, acc3 = {};

    // A-fragment source: row = rowbase + (lane&15); K chunks at g*8 and 16+g*8
    const float* arow = x + (size_t)(rowbase + (lane & 15)) * D_DIM + (lane >> 4) * 8;

    auto loadA = [&](int t) -> v16bf {
        const float* ap = arow + t * 32;
        float4 c0 = *(const float4*)(ap);
        float4 c1 = *(const float4*)(ap + 4);
        float4 c2 = *(const float4*)(ap + 16);
        float4 c3 = *(const float4*)(ap + 20);
        v16bf a;   // native casts -> hardware bf16 cvt (pk form when paired)
        a[0]  = (__bf16)c0.x; a[1]  = (__bf16)c0.y; a[2]  = (__bf16)c0.z; a[3]  = (__bf16)c0.w;
        a[4]  = (__bf16)c1.x; a[5]  = (__bf16)c1.y; a[6]  = (__bf16)c1.z; a[7]  = (__bf16)c1.w;
        a[8]  = (__bf16)c2.x; a[9]  = (__bf16)c2.y; a[10] = (__bf16)c2.z; a[11] = (__bf16)c2.w;
        a[12] = (__bf16)c3.x; a[13] = (__bf16)c3.y; a[14] = (__bf16)c3.z; a[15] = (__bf16)c3.w;
        return a;
    };

    // distance-2 software pipeline on the A stream (two loads in flight)
    v16bf a0 = loadA(0);
    v16bf a1 = loadA(1);
    for (int t = 0; t < K_STEPS; ++t) {
        __builtin_prefetch(arow + t * 32 + 256, 0, 1);   // warm L2 8 k-steps ahead
        const int tn = (t + 2 < K_STEPS) ? t + 2 : K_STEPS - 1;
        v16bf a2 = loadA(tn);

        const unsigned short* bbase = smemB + ((size_t)(t * 4) * 32 + lane) * 16;
        v16bf b0 = *(const v16bf*)(bbase);
        v16bf b1 = *(const v16bf*)(bbase + 32 * 16);
        v16bf b2 = *(const v16bf*)(bbase + 2 * 32 * 16);
        v16bf b3 = *(const v16bf*)(bbase + 3 * 32 * 16);

        acc0 = __builtin_amdgcn_wmma_f32_16x16x32_bf16(false, a0, false, b0, (short)0, acc0, false, false);
        acc1 = __builtin_amdgcn_wmma_f32_16x16x32_bf16(false, a0, false, b1, (short)0, acc1, false, false);
        acc2 = __builtin_amdgcn_wmma_f32_16x16x32_bf16(false, a0, false, b2, (short)0, acc2, false, false);
        acc3 = __builtin_amdgcn_wmma_f32_16x16x32_bf16(false, a0, false, b3, (short)0, acc3, false, false);

        a0 = a1;
        a1 = a2;
    }

    // ---- Epilogue: per-row top-2 + gates + scatter ----
    float* out_combine = out;
    float* out_idx     = out + (size_t)T_TOK * N_EXP;
    float* out_gates   = out_idx + (size_t)T_TOK * 2;
    float* out_act     = out_gates + (size_t)T_TOK * 2;

    const int half  = lane >> 4;          // 0: rows v, 1: rows v+8
    const int cbase = lane & 15;          // column-within-16-block owned by lane

#pragma unroll
    for (int v = 0; v < 8; ++v) {
        const int row = rowbase + v + half * 8;
        const float* nrow = noise + (size_t)row * N_EXP;

        // C layout: VGPR v, lanes 0-15 -> M=v, lanes 16-31 -> M=v+8, N = cbase + nb*16
        float s0 = acc0[v] + nrow[cbase];
        float s1 = acc1[v] + nrow[16 + cbase];
        float s2 = acc2[v] + nrow[32 + cbase];
        float s3 = acc3[v] + nrow[48 + cbase];

        // local top-2 of 4 (value, column)
        float v1, v2; int i1, i2;
        if (s0 >= s1) { v1 = s0; i1 = cbase;      v2 = s1; i2 = 16 + cbase; }
        else          { v1 = s1; i1 = 16 + cbase; v2 = s0; i2 = cbase; }
        if (s2 > v1)      { v2 = v1; i2 = i1; v1 = s2; i1 = 32 + cbase; }
        else if (s2 > v2) { v2 = s2; i2 = 32 + cbase; }
        if (s3 > v1)      { v2 = v1; i2 = i1; v1 = s3; i1 = 48 + cbase; }
        else if (s3 > v2) { v2 = s3; i2 = 48 + cbase; }

        // butterfly merge across the 16 lanes of this half (xor < 16 stays in half)
#pragma unroll
        for (int m = 1; m < 16; m <<= 1) {
            float ov1 = __shfl_xor(v1, m, 32);
            int   oi1 = __shfl_xor(i1, m, 32);
            float ov2 = __shfl_xor(v2, m, 32);
            int   oi2 = __shfl_xor(i2, m, 32);
            if (ov1 > v1) {
                float tv = v1; int ti = i1;
                v1 = ov1; i1 = oi1;
                if (ov2 > tv) { v2 = ov2; i2 = oi2; }
                else          { v2 = tv;  i2 = ti; }
            } else if (ov1 > v2) {
                v2 = ov1; i2 = oi1;
            }
        }

        // renormalized gates: e^{s1}/(e^{s1}+e^{s2}) = 1/(1+e^{s2-s1}), s2<=s1 -> stable
        float g1 = 1.0f / (1.0f + __expf(v2 - v1));
        float g2 = 1.0f - g1;

        // scatter combine row (write all 64 entries; out buffer is poisoned)
        float* crow = out_combine + (size_t)row * N_EXP;
#pragma unroll
        for (int nb = 0; nb < 4; ++nb) {
            int col = nb * 16 + cbase;
            float val = (col == i1) ? g1 : ((col == i2) ? g2 : 0.0f);
            crow[col] = val;
        }

        if (cbase == 0) {
            out_idx[(size_t)row * 2 + 0]   = (float)i1;
            out_idx[(size_t)row * 2 + 1]   = (float)i2;
            out_gates[(size_t)row * 2 + 0] = g1;
            out_gates[(size_t)row * 2 + 1] = g2;
            atomicAdd(&out_act[i1], 1.0f);
            atomicAdd(&out_act[i2], 1.0f);
        }
    }
}

extern "C" void kernel_launch(void* const* d_in, const int* in_sizes, int n_in,
                              void* d_out, int out_size, void* d_ws, size_t ws_size,
                              hipStream_t stream) {
    (void)in_sizes; (void)n_in; (void)out_size; (void)ws_size;
    const float* x      = (const float*)d_in[0];
    const float* w_gate = (const float*)d_in[1];
    const float* noise  = (const float*)d_in[2];
    float* out = (float*)d_out;
    unsigned short* packedB = (unsigned short*)d_ws;   // 2048*64 bf16 = 256 KB

    float* out_act = out + (size_t)T_TOK * N_EXP + (size_t)T_TOK * 2 * 2;

    // pack B fragments (+ zero activation counters) : 131072 elements
    router_pack_b<<<512, 256, 0, stream>>>(w_gate, packedB, out_act);

    // fused GEMM + router: 256 rows per block, 16 waves, 256 KB dynamic LDS for B
    router_fused<<<T_TOK / ROWS_PER_BLOCK, BLOCK_THREADS, B_BYTES, stream>>>(x, packedB, noise, out);
}
